// NonLocalBlock_84825604096328
// MI455X (gfx1250) — matrix operand order repaired and checked by reading, not compiled, not run on hardware
//
#include <hip/hip_runtime.h>

// Problem sizes (fixed by the reference): B=8, C=192, LATENT=96, N=64*64
#define BATCH 8
#define CCH   192
#define LAT   96
#define NPIX  4096
// 1/sqrt(96) * log2(e): folded into Q at projection so the softmax runs in the
// log2 domain and each exponential is a bare v_sub + v_exp_f32.
#define QSCALE 0.147244697452490f

typedef __bf16 bf16_t;
typedef __attribute__((ext_vector_type(16))) __bf16 v16bf;
typedef __attribute__((ext_vector_type(2)))  __bf16 v2bf;
typedef __attribute__((ext_vector_type(8)))  float  v8f;
typedef __attribute__((ext_vector_type(4)))  int    i4;

union TileU { i4 q[2]; v16bf v; };
union WU   { unsigned u[8]; v16bf v; };

// A-fragment (16x32 bf16): lane(h=lane>>4): K = {8h..8h+7} and {16+8h..16+8h+7}
// -> two contiguous 16B loads at +0 and +16 elements (base already has +8h).
__device__ __forceinline__ v16bf load_tile_a(const bf16_t* p) {
  TileU t;
  t.q[0] = *(const i4*)(p);
  t.q[1] = *(const i4*)(p + 16);
  return t.v;
}
// B-fragment (32x16 bf16): lane holds K = {8h..8h+7} and {16+8h..16+8h+7} of its column
// -> two contiguous 16B loads at +0 and +16 elements (base already has +8h).
__device__ __forceinline__ v16bf load_tile_b(const bf16_t* p) {
  TileU t;
  t.q[0] = *(const i4*)(p);
  t.q[1] = *(const i4*)(p + 16);
  return t.v;
}
// Packed f32x2 -> bf16x2 conversion: selects a single v_cvt_pk_bf16_f32.
__device__ __forceinline__ unsigned pack_bf16(float a, float b) {
  v2bf p;
  p[0] = (__bf16)a;
  p[1] = (__bf16)b;
  return __builtin_bit_cast(unsigned, p);
}

// Online-softmax update (log2 domain) for one 16-column group + pack P into the
// B-operand layout of the V@P WMMA. s0v/s1v: S rows q=8h+i and q=16+8h+i of
// this lane's column. Rescale of acc/l is skipped (exactly) when no column's
// max moved (wave-uniform ballot branch).
__device__ __forceinline__ v16bf softmax_pack(v8f& s0v, v8f& s1v,
                                              float& m_run, float& l_run,
                                              v8f* acc, int h) {
  float tmax = -3.0e38f;
#pragma unroll
  for (int i = 0; i < 8; ++i) tmax = fmaxf(tmax, fmaxf(s0v[i], s1v[i]));
  tmax = fmaxf(tmax, __shfl_xor(tmax, 16, 32));
  if (__ballot(tmax > m_run)) {      // rarely taken after warmup
    float mnew = fmaxf(m_run, tmax);
    float corr = __builtin_amdgcn_exp2f(m_run - mnew);
    m_run = mnew;
    l_run *= corr;
#pragma unroll
    for (int t = 0; t < 6; ++t)
#pragma unroll
      for (int i = 0; i < 8; ++i) acc[t][i] *= corr;
  }
  float p0[8], p1[8], psum = 0.0f;
#pragma unroll
  for (int i = 0; i < 8; ++i) {
    p0[i] = __builtin_amdgcn_exp2f(s0v[i] - m_run);
    p1[i] = __builtin_amdgcn_exp2f(s1v[i] - m_run);
    psum += p0[i] + p1[i];
  }
  l_run += psum;                     // per-lane partial; combined at the end
  unsigned pk0[4], pk1[4];
#pragma unroll
  for (int j = 0; j < 4; ++j) {
    pk0[j] = pack_bf16(p0[2 * j], p0[2 * j + 1]);
    pk1[j] = pack_bf16(p1[2 * j], p1[2 * j + 1]);
  }
  WU wb;
#pragma unroll
  for (int j = 0; j < 4; ++j) {
    wb.u[j]     = pk0[j];
    wb.u[4 + j] = pk1[j];
  }
  return wb.v;
}

// ---------------- Kernel 1: T = Wt @ X + bt, routed into Qt / V / Kt (bf16) ---
// 16 outputs per thread: x column read once per 16 o's, Wt reads are
// wave-uniform (scalar loads). Q is pre-scaled by log2(e)/sqrt(96).
__global__ __launch_bounds__(256)
void proj_qkv(const float* __restrict__ x, const float* __restrict__ Wt,
              const float* __restrict__ bt,
              bf16_t* __restrict__ Qt, bf16_t* __restrict__ Kt,
              bf16_t* __restrict__ Vm) {
  int n  = blockIdx.x * 256 + threadIdx.x;
  int ob = blockIdx.y * 16;
  int b  = blockIdx.z;
  const float* xb = x + (size_t)b * CCH * NPIX + n;
  float acc[16];
#pragma unroll
  for (int j = 0; j < 16; ++j) acc[j] = bt[ob + j];
#pragma unroll 4
  for (int c = 0; c < CCH; ++c) {
    float xv = xb[(size_t)c * NPIX];
#pragma unroll
    for (int j = 0; j < 16; ++j)
      acc[j] = fmaf(Wt[(size_t)(ob + j) * CCH + c], xv, acc[j]);
  }
  if (ob < LAT) {
#pragma unroll
    for (int j = 0; j < 16; ++j)
      Qt[((size_t)b * NPIX + n) * LAT + ob + j] = (bf16_t)(acc[j] * QSCALE);
  } else if (ob < 2 * LAT) {
#pragma unroll
    for (int j = 0; j < 16; ++j)
      Vm[((size_t)b * LAT + (ob - LAT + j)) * NPIX + n] = (bf16_t)acc[j];
  } else {
#pragma unroll
    for (int j = 0; j < 16; ++j)
      Kt[((size_t)b * NPIX + n) * LAT + (ob - 2 * LAT + j)] = (bf16_t)acc[j];
  }
}

// ---------------- Kernel 2: fused flash attention (softmax over q) ------------
// One wave32 owns batch b and 32 output columns (two 16-column groups sharing
// all Q/V A-fragments -> 24 WMMAs per 12KB of loads).
__global__ __launch_bounds__(256)
void attn_fused(const bf16_t* __restrict__ Qt, const bf16_t* __restrict__ Kt,
                const bf16_t* __restrict__ Vm, float* __restrict__ O) {
  int gwave = (blockIdx.x * 256 + (int)threadIdx.x) >> 5;  // 0..1023
  int lane  = threadIdx.x & 31;
  int b     = gwave >> 7;              // 128 column-blocks of 32 per batch
  int kbase = (gwave & 127) << 5;
  int h  = lane >> 4;
  int mr = lane & 15;

  // Resident B-fragments of K for both column groups (c = 0..95, 3 chunks)
  const bf16_t* kRow0 = Kt + ((size_t)b * NPIX + kbase + mr) * LAT + 8 * h;
  const bf16_t* kRow1 = kRow0 + (size_t)16 * LAT;
  v16bf bK0[3], bK1[3];
#pragma unroll
  for (int t = 0; t < 3; ++t) {
    bK0[t] = load_tile_b(kRow0 + 32 * t);
    bK1[t] = load_tile_b(kRow1 + 32 * t);
  }

  const bf16_t* qPtr = Qt + ((size_t)b * NPIX + mr) * LAT + 8 * h;
  const bf16_t* vPtr = Vm + ((size_t)b * LAT + mr) * NPIX + 8 * h;

  v8f acc0[6], acc1[6];
#pragma unroll
  for (int t = 0; t < 6; ++t)
#pragma unroll
    for (int i = 0; i < 8; ++i) { acc0[t][i] = 0.0f; acc1[t][i] = 0.0f; }

  float m0 = -3.0e38f, l0 = 0.0f;
  float m1 = -3.0e38f, l1 = 0.0f;

  for (int qb = 0; qb < NPIX; qb += 32) {
    const bf16_t* q0 = qPtr + (size_t)qb * LAT;
    const bf16_t* q1 = q0 + 16 * LAT;
    v8f s00 = {}, s01 = {}, s10 = {}, s11 = {};
#pragma unroll
    for (int t = 0; t < 3; ++t) {
      v16bf a0 = load_tile_a(q0 + 32 * t);
      v16bf a1 = load_tile_a(q1 + 32 * t);
      s00 = __builtin_amdgcn_wmma_f32_16x16x32_bf16(false, a0, false, bK0[t],
                                                    (short)0, s00, false, false);
      s01 = __builtin_amdgcn_wmma_f32_16x16x32_bf16(false, a1, false, bK0[t],
                                                    (short)0, s01, false, false);
      s10 = __builtin_amdgcn_wmma_f32_16x16x32_bf16(false, a0, false, bK1[t],
                                                    (short)0, s10, false, false);
      s11 = __builtin_amdgcn_wmma_f32_16x16x32_bf16(false, a1, false, bK1[t],
                                                    (short)0, s11, false, false);
    }

    v16bf bP0 = softmax_pack(s00, s01, m0, l0, acc0, h);
    v16bf bP1 = softmax_pack(s10, s11, m1, l1, acc1, h);

#pragma unroll
    for (int t = 0; t < 6; ++t) {
      v16bf aV = load_tile_a(vPtr + (size_t)t * 16 * NPIX + qb);
      acc0[t] = __builtin_amdgcn_wmma_f32_16x16x32_bf16(false, aV, false, bP0,
                                                        (short)0, acc0[t], false, false);
      acc1[t] = __builtin_amdgcn_wmma_f32_16x16x32_bf16(false, aV, false, bP1,
                                                        (short)0, acc1[t], false, false);
    }
  }

  float lt0 = l0 + __shfl_xor(l0, 16, 32);
  float lt1 = l1 + __shfl_xor(l1, 16, 32);
  float inv0 = __builtin_amdgcn_rcpf(lt0);
  float inv1 = __builtin_amdgcn_rcpf(lt1);
  float* oc0 = O + (size_t)b * LAT * NPIX + kbase + mr;
  float* oc1 = oc0 + 16;
#pragma unroll
  for (int t = 0; t < 6; ++t)
#pragma unroll
    for (int i = 0; i < 8; ++i) {
      size_t row = (size_t)(t * 16 + 8 * h + i) * NPIX;
      oc0[row] = acc0[t][i] * inv0;
      oc1[row] = acc1[t][i] * inv1;
    }
}

// ---------------- Kernel 3: out = Wd @ O + bd + x -----------------------------
__global__ __launch_bounds__(256)
void proj_out(const float* __restrict__ O, const float* __restrict__ x,
              const float* __restrict__ Wd, const float* __restrict__ bd,
              float* __restrict__ out) {
  int n  = blockIdx.x * 256 + threadIdx.x;
  int ob = blockIdx.y * 16;
  int b  = blockIdx.z;
  const float* obp = O + (size_t)b * LAT * NPIX + n;
  float acc[16];
#pragma unroll
  for (int j = 0; j < 16; ++j) acc[j] = bd[ob + j];
#pragma unroll 4
  for (int c = 0; c < LAT; ++c) {
    float ov = obp[(size_t)c * NPIX];
#pragma unroll
    for (int j = 0; j < 16; ++j)
      acc[j] = fmaf(Wd[(size_t)(ob + j) * LAT + c], ov, acc[j]);
  }
#pragma unroll
  for (int j = 0; j < 16; ++j) {
    size_t idx = ((size_t)b * CCH + ob + j) * NPIX + n;
    out[idx] = acc[j] + x[idx];
  }
}

extern "C" void kernel_launch(void* const* d_in, const int* in_sizes, int n_in,
                              void* d_out, int out_size, void* d_ws, size_t ws_size,
                              hipStream_t stream) {
  (void)in_sizes; (void)n_in; (void)out_size; (void)ws_size;
  const float* x  = (const float*)d_in[0];
  const float* Wt = (const float*)d_in[1];
  const float* bt = (const float*)d_in[2];
  const float* Wd = (const float*)d_in[3];
  const float* bd = (const float*)d_in[4];
  float* out = (float*)d_out;

  // Workspace layout (needs ~31.5 MB):
  //   Qt : B*N*LAT bf16   (q-major, c contiguous, pre-scaled by log2e/sqrt(96))
  //   Kt : B*N*LAT bf16   (k-major, c contiguous)
  //   Vm : B*LAT*N bf16   (c-major, n contiguous)
  //   O  : B*LAT*N f32    (attention output before back-projection)
  bf16_t* Qt = (bf16_t*)d_ws;
  bf16_t* Kt = Qt + (size_t)BATCH * NPIX * LAT;
  bf16_t* Vm = Kt + (size_t)BATCH * NPIX * LAT;
  float*  Ob = (float*)(Vm + (size_t)BATCH * LAT * NPIX);

  proj_qkv<<<dim3(NPIX / 256, (3 * LAT) / 16, BATCH), 256, 0, stream>>>(x, Wt, bt, Qt, Kt, Vm);
  attn_fused<<<dim3((BATCH * (NPIX / 32)) / 8), 256, 0, stream>>>(Qt, Kt, Vm, Ob);
  proj_out<<<dim3(NPIX / 256, CCH / 16, BATCH), 256, 0, stream>>>(Ob, x, Wd, bd, out);
}
